// TicMIL_Parallel_Head_28836410426006
// MI455X (gfx1250) — matrix operand order repaired and checked by reading, not compiled, not run on hardware
//
#include <hip/hip_runtime.h>

#define KCLUS 3
#define KMITERS 50
#define BAGSZ 1042
#define NCL 961
#define DIMS 768
#define NTILES 61      // ceil(961/16)
#define KSTEPS 24      // 768 / 32
#define TPB 768        // 24 wave32 per bag -> 6 waves per SIMD32
#define NWAVES (TPB / 32)
#define NBACK (BAGSZ - NCL)        // 81
#define BACK_ELEMS (NBACK * DIMS)  // 62208

typedef __attribute__((ext_vector_type(16))) _Float16 v16h;
typedef __attribute__((ext_vector_type(8)))  _Float16 v8h;
typedef __attribute__((ext_vector_type(8)))  float    v8f;

__device__ __forceinline__ float rnd01(unsigned x) {
    x ^= x >> 16; x *= 0x7feb352dU;
    x ^= x >> 15; x *= 0x846ca68bU;
    x ^= x >> 16;
    return (float)(x >> 8) * (1.0f / 16777216.0f);
}

// One-time f32 -> f16 mirror of x (row-major, same layout).
__global__ void __launch_bounds__(256)
cvt_f16_kernel(const float* __restrict__ x, _Float16* __restrict__ xh)
{
    size_t i = ((size_t)blockIdx.x * 256 + threadIdx.x) * 8;
    float4 a = *(const float4*)(x + i);
    float4 b = *(const float4*)(x + i + 4);
    v8h h;
    h[0] = (_Float16)a.x; h[1] = (_Float16)a.y; h[2] = (_Float16)a.z; h[3] = (_Float16)a.w;
    h[4] = (_Float16)b.x; h[5] = (_Float16)b.y; h[6] = (_Float16)b.z; h[7] = (_Float16)b.w;
    *(v8h*)(xh + i) = h;
}

// One workgroup (24 wave32) per bag: full k-means + bag statistics.
template <bool USE_F16>
__global__ void __launch_bounds__(TPB)
kmeans_bag_kernel(const float* __restrict__ x,
                  const _Float16* __restrict__ xh,  // f16 mirror (may be null when !USE_F16)
                  float* __restrict__ bm_out,       // [B, DIMS]
                  float* __restrict__ mn_out,       // [B]
                  float* __restrict__ nm_out)       // [B]
{
    __shared__ float centersS[KCLUS][DIMS];       // 9216 B
    __shared__ v16h  BldsS[KSTEPS][32];           // 24576 B, WMMA-ready B operand
    __shared__ float cnS[KCLUS];                  // ||c_k||^2
    __shared__ unsigned char assignS[NTILES * 16];
    __shared__ float rowsumS[KCLUS];
    __shared__ float backsumS;
    __shared__ int   cntFinS[KCLUS];
    __shared__ float scaleS[KCLUS];

    const int tid  = threadIdx.x;                 // == owned dim (0..767)
    const int lane = tid & 31;
    const int wave = tid >> 5;
    const int bag  = blockIdx.x;

    const float* xb  = x + (size_t)bag * BAGSZ * DIMS;
    const float* col = xb + tid;                  // this thread's owned-dim column base
    const _Float16* xbh = USE_F16 ? (xh + (size_t)bag * BAGSZ * DIMS) : nullptr;

    // ---------- center init: per-feature [min,max] + deterministic uniform ----------
    {
        float lo = col[0], hi = lo;
        for (int i = 1; i < NCL; ++i) {
            float v = col[(size_t)i * DIMS];
            lo = fminf(lo, v); hi = fmaxf(hi, v);
        }
        for (int k = 0; k < KCLUS; ++k) {
            unsigned seed = (unsigned)(bag * KCLUS * DIMS + k * DIMS + tid) + 0x9E3779B9u;
            centersS[k][tid] = lo + rnd01(seed) * (hi - lo);
        }
    }
    __syncthreads();

    // ---------- k-means: 50 update iterations + 1 final assignment pass ----------
    for (int it = 0; it <= KMITERS; ++it) {
        // --- build phase: pack B operand (centers -> f16 swizzled) + ||c_k||^2 ---
        if (tid < KCLUS) cnS[tid] = 0.f;
        __syncthreads();

        {   // one B entry per thread: 768 == KSTEPS*32
            int ks = tid >> 5, ln = tid & 31, n = ln & 15;
            // B (K=32 x N=16) f16 layout: lanes 0-15 hold K{0..7,16..23}, lanes 16-31 hold K{8..15,24..31}
            int kb = ks * 32 + ((ln < 16) ? 0 : 8);
            v16h bv;
            if (n < KCLUS) {
#pragma unroll
                for (int j = 0; j < 16; ++j) bv[j] = (_Float16)centersS[n][kb + (j < 8 ? j : j + 8)];
            } else {
#pragma unroll
                for (int j = 0; j < 16; ++j) bv[j] = (_Float16)0.f;
            }
            BldsS[ks][ln] = bv;
        }
        {
            float c0 = centersS[0][tid], c1 = centersS[1][tid], c2 = centersS[2][tid];
            atomicAdd(&cnS[0], c0 * c0);
            atomicAdd(&cnS[1], c1 * c1);
            atomicAdd(&cnS[2], c2 * c2);
        }
        __syncthreads();

        // --- assignment phase: WMMA f32<-f16 GEMM X(16x768) x C^T(768x16) per tile ---
        const float cn0 = cnS[0], cn1 = cnS[1], cn2 = cnS[2];
        for (int tile = wave; tile < NTILES; tile += NWAVES) {
            const int rbase = tile * 16;
            const int row = rbase + (lane & 15);       // A: M = lane&15
            // rows >= 961 read 'back' rows: valid memory, outputs discarded
            const int o0 = (lane < 16) ? 0 : 8;        // A f16 16x32: low lanes K{0..7,16..23}
            const int o1 = (lane < 16) ? 16 : 24;      //              high lanes K{8..15,24..31}

            // Opaque zero, re-defined every tile iteration: prevents LICM from
            // hoisting all 24 LDS B-operand reads out of the loop (which caused
            // a 192-VGPR spill to scratch).  Keeps B as in-loop ds_load_b128.
            int ksbase = 0;
            asm volatile("" : "+v"(ksbase));

            v8f acc0 = {}, acc1 = {};   // two accumulators: break the WMMA RAW chain
            if constexpr (USE_F16) {
                const _Float16* xr = xbh + (size_t)row * DIMS;
                for (int ks = 0; ks < KSTEPS; ++ks) {
                    const _Float16* p = xr + ks * 32;
                    v8h h0 = *(const v8h*)(p + o0);
                    v8h h1 = *(const v8h*)(p + o1);
                    v16h a;
#pragma unroll
                    for (int j = 0; j < 8; ++j) { a[j] = h0[j]; a[8 + j] = h1[j]; }
                    v16h b = BldsS[ks + ksbase][lane];
                    if (ks & 1)
                        acc1 = __builtin_amdgcn_wmma_f32_16x16x32_f16(
                            false, a, false, b, (short)0, acc1, false, false);
                    else
                        acc0 = __builtin_amdgcn_wmma_f32_16x16x32_f16(
                            false, a, false, b, (short)0, acc0, false, false);
                }
            } else {
                const float* xr = xb + (size_t)row * DIMS;
                for (int ks = 0; ks < KSTEPS; ++ks) {
                    const float* p = xr + ks * 32;
                    const float4 f0 = *(const float4*)(p + o0);
                    const float4 f1 = *(const float4*)(p + o0 + 4);
                    const float4 f2 = *(const float4*)(p + o1);
                    const float4 f3 = *(const float4*)(p + o1 + 4);
                    v16h a;
                    a[0]  = (_Float16)f0.x; a[1]  = (_Float16)f0.y; a[2]  = (_Float16)f0.z; a[3]  = (_Float16)f0.w;
                    a[4]  = (_Float16)f1.x; a[5]  = (_Float16)f1.y; a[6]  = (_Float16)f1.z; a[7]  = (_Float16)f1.w;
                    a[8]  = (_Float16)f2.x; a[9]  = (_Float16)f2.y; a[10] = (_Float16)f2.z; a[11] = (_Float16)f2.w;
                    a[12] = (_Float16)f3.x; a[13] = (_Float16)f3.y; a[14] = (_Float16)f3.z; a[15] = (_Float16)f3.w;
                    v16h b = BldsS[ks + ksbase][lane];
                    if (ks & 1)
                        acc1 = __builtin_amdgcn_wmma_f32_16x16x32_f16(
                            false, a, false, b, (short)0, acc1, false, false);
                    else
                        acc0 = __builtin_amdgcn_wmma_f32_16x16x32_f16(
                            false, a, false, b, (short)0, acc0, false, false);
                }
            }
            // C/D: VGPR r -> lanes0-15 M=r, lanes16-31 M=8+r; N = lane&15 (clusters at N=0..2)
#pragma unroll
            for (int r = 0; r < 8; ++r) {
                float v = acc0[r] + acc1[r];
                float e00 = __shfl(v, 0, 32),  e01 = __shfl(v, 1, 32),  e02 = __shfl(v, 2, 32);
                float e10 = __shfl(v, 16, 32), e11 = __shfl(v, 17, 32), e12 = __shfl(v, 18, 32);
                if (lane == 0) {
                    int rw = rbase + r;
                    if (rw < NCL) {
                        float s0 = cn0 - 2.f * e00, s1 = cn1 - 2.f * e01, s2 = cn2 - 2.f * e02;
                        int a8 = 0; float bst = s0;
                        if (s1 < bst) { bst = s1; a8 = 1; }
                        if (s2 < bst) { a8 = 2; }
                        assignS[rw] = (unsigned char)a8;
                    }
                } else if (lane == 1) {
                    int rw = rbase + 8 + r;
                    if (rw < NCL) {
                        float s0 = cn0 - 2.f * e10, s1 = cn1 - 2.f * e11, s2 = cn2 - 2.f * e12;
                        int a8 = 0; float bst = s0;
                        if (s1 < bst) { bst = s1; a8 = 1; }
                        if (s2 < bst) { a8 = 2; }
                        assignS[rw] = (unsigned char)a8;
                    }
                }
            }
        }
        __syncthreads();

        if (it == KMITERS) break;   // uniform

        // --- update phase: per-cluster sums over owned dim (assign is block-uniform) ---
        {
            float s0 = 0, s1 = 0, s2 = 0;
            int c0 = 0, c1 = 0, c2 = 0;
            for (int i = 0; i < NCL; ++i) {
                int ai = assignS[i];                 // same value for every thread -> scalar branch
                float v = col[(size_t)i * DIMS];
                if (ai == 0)      { s0 += v; c0++; }
                else if (ai == 1) { s1 += v; c1++; }
                else              { s2 += v; c2++; }
            }
            if (c0 > 0) centersS[0][tid] = s0 / (float)c0;
            if (c1 > 0) centersS[1][tid] = s1 / (float)c1;
            if (c2 > 0) centersS[2][tid] = s2 / (float)c2;
        }
        __syncthreads();
    }

    // ---------- bag statistics with final assignment ----------
    if (tid < KCLUS) rowsumS[tid] = 0.f;
    if (tid == KCLUS) backsumS = 0.f;
    __syncthreads();
    {
        float s0 = 0, s1 = 0, s2 = 0; int c0 = 0, c1 = 0, c2 = 0;
        for (int i = 0; i < NCL; ++i) {
            int ai = assignS[i];
            float v = col[(size_t)i * DIMS];
            if (ai == 0)      { s0 += v; c0++; }
            else if (ai == 1) { s1 += v; c1++; }
            else              { s2 += v; c2++; }
        }
        atomicAdd(&rowsumS[0], s0); atomicAdd(&rowsumS[1], s1); atomicAdd(&rowsumS[2], s2);
        if (tid == 0) { cntFinS[0] = c0; cntFinS[1] = c1; cntFinS[2] = c2; }

        const float* back = xb + (size_t)NCL * DIMS;
        float bs = 0.f;
        for (int e = tid; e < BACK_ELEMS; e += TPB) bs += back[e];
        atomicAdd(&backsumS, bs);
    }
    __syncthreads();
    if (tid == 0) {
        float mt = backsumS / (float)BACK_ELEMS;
        float mn = 1e30f, sm = 0.f;
        for (int k = 0; k < KCLUS; ++k) {
            int c = cntFinS[k];
            float cm = (c > 0) ? rowsumS[k] / ((float)c * (float)DIMS) : 0.f;
            float ds = fabsf(mt - cm);
            scaleS[k] = 1.f - ds;
            sm += ds; mn = fminf(mn, ds);
        }
        mn_out[bag] = mn;
        nm_out[bag] = sm - mn;
    }
    __syncthreads();
    {
        float b0 = 0.f;
        for (int i = 0; i < NCL; ++i)
            b0 += col[(size_t)i * DIMS] * scaleS[assignS[i]];
        for (int j = 0; j < NBACK; ++j)
            b0 += col[(size_t)(NCL + j) * DIMS];
        bm_out[(size_t)bag * DIMS + tid] = b0 * (1.f / (float)BAGSZ);
    }
}

// Tiny head GEMM (32x3, K=768) + scalar means.
__global__ void __launch_bounds__(128)
finalize_kernel(const float* __restrict__ bm, const float* __restrict__ mn,
                const float* __restrict__ nm, const float* __restrict__ hw,
                const float* __restrict__ hb, float* __restrict__ out, int B)
{
    int t = threadIdx.x;
    if (t < 3 * B) {
        int b = t / 3, c = t % 3;
        const float* v = bm + (size_t)b * DIMS;
        const float* w = hw + (size_t)c * DIMS;
        float s = hb[c];
        for (int d = 0; d < DIMS; ++d) s += v[d] * w[d];
        out[t] = s;
    } else if (t == 3 * B) {
        float s = 0.f;
        for (int b = 0; b < B; ++b) s += mn[b];
        out[3 * B] = s / (float)B;
    } else if (t == 3 * B + 1) {
        float s = 0.f;
        for (int b = 0; b < B; ++b) s += nm[b];
        out[3 * B + 1] = s / (float)B;
    }
}

extern "C" void kernel_launch(void* const* d_in, const int* in_sizes, int n_in,
                              void* d_out, int out_size, void* d_ws, size_t ws_size,
                              hipStream_t stream) {
    const float* x  = (const float*)d_in[0];
    const float* hw = (const float*)d_in[1];
    const float* hb = (const float*)d_in[2];
    float* out = (float*)d_out;

    const int B = in_sizes[0] / (BAGSZ * DIMS);   // 32
    const size_t nelem = (size_t)B * BAGSZ * DIMS;

    float* wsf = (float*)d_ws;
    float* bm = wsf;                   // B * DIMS floats
    float* mn = bm + (size_t)B * DIMS; // B floats
    float* nm = mn + B;                // B floats
    const size_t base_bytes = ((size_t)B * DIMS + 2 * (size_t)B) * sizeof(float);

    // f16 mirror of x placed after the float outputs (16B-aligned offset).
    const size_t need_f16 = base_bytes + nelem * sizeof(_Float16);

    if (ws_size >= need_f16) {
        _Float16* xh = (_Float16*)((char*)d_ws + base_bytes);
        const int cvt_blocks = (int)(nelem / (256 * 8));   // nelem divisible by 2048
        cvt_f16_kernel<<<cvt_blocks, 256, 0, stream>>>(x, xh);
        kmeans_bag_kernel<true><<<B, TPB, 0, stream>>>(x, xh, bm, mn, nm);
    } else {
        kmeans_bag_kernel<false><<<B, TPB, 0, stream>>>(x, nullptr, bm, mn, nm);
    }
    finalize_kernel<<<1, 128, 0, stream>>>(bm, mn, nm, hw, hb, out, B);
}
